// NeuralODEWrapper_77515569758860
// MI455X (gfx1250) — compile-verified, matrix operand order checked by reading, not run on hardware
//
#include <hip/hip_runtime.h>
#include <hip/hip_bf16.h>

// CDNA5 / gfx1250 persistent Neural-ODE RK4 rollout.
// Hidden layers: D = A(16x32 f16) x B(32x16 f16) + C(16x16 f32)
// via v_wmma_f32_16x16x32_f16; weights register-resident, activations in LDS.

typedef _Float16 h8   __attribute__((ext_vector_type(8)));
typedef _Float16 v16h __attribute__((ext_vector_type(16)));
typedef float    v8f  __attribute__((ext_vector_type(8)));

#define HIDDEN   256
#define NLAYER   3          // hidden 256x256 layers
#define MT       16         // batch rows per workgroup (one WMMA M-tile)
#define THREADS  512        // 16 waves, one 16-col N-tile per wave
#define HPAD     (HIDDEN + 8)
#define DT       0.005f

// Branch-free tanh: native v_tanh_f32 on gfx1250, else exp-based rational.
static __device__ __forceinline__ float fast_tanh(float x) {
#if __has_builtin(__builtin_amdgcn_tanhf)
    return __builtin_amdgcn_tanhf(x);
#else
    const float e = __expf(2.0f * x);                 // v_exp_f32 path
    return (e - 1.0f) * __builtin_amdgcn_rcpf(e + 1.0f);
#endif
}

__launch_bounds__(THREADS, 1)
__global__ void node_rollout_kernel(const float* __restrict__ z0,
                                    const float* __restrict__ t0,
                                    const float* __restrict__ W_in,
                                    const float* __restrict__ b_in,
                                    const float* __restrict__ W_h,
                                    const float* __restrict__ b_h,
                                    const float* __restrict__ W_out,
                                    const float* __restrict__ b_out,
                                    const int*   __restrict__ steps_p,
                                    float*       __restrict__ out)
{
    __shared__ _Float16 hbuf[2][MT][HPAD];   // ping-pong activations (f16)
    __shared__ float    xbuf[MT][5];         // vf input features
    __shared__ float    kbuf[MT][2];         // stage derivative k

    const int tid  = threadIdx.x;
    const int lane = tid & 31;
    const int wave = tid >> 5;               // 0..15, owns cols [wave*16, wave*16+16)
    const int row0 = blockIdx.x * MT;
    const int steps = *steps_p;

    const int nloc = lane & 15;              // column within tile / A-row per lane
    const int g    = lane >> 4;              // half-wave group
    const int col  = wave * 16 + nloc;       // global hidden unit owned by this lane

    // ---------------- one-time register-resident weight fragments ----------------
    // B-matrix (32x16 f16) fragment layout, mirrored from the documented 16-bit
    // A layout: lanes 0-15 hold n=lane, lanes 16-31 hold n=lane-16;
    // VGPR v<4: k = 2v+e + 8g ; v>=4: k = 16 + 2(v-4)+e + 8g.
    v16h wfrag[NLAYER][8];
    {
        #pragma unroll
        for (int l = 0; l < NLAYER; ++l) {
            const float* Wl = W_h + (size_t)l * HIDDEN * HIDDEN + (size_t)col * HIDDEN;
            #pragma unroll
            for (int kf = 0; kf < 8; ++kf) {
                v16h f;
                #pragma unroll
                for (int v = 0; v < 8; ++v) {
                    #pragma unroll
                    for (int e = 0; e < 2; ++e) {
                        const int k = kf * 32 +
                            (v < 4 ? (v * 2 + e + g * 8)
                                   : (16 + (v - 4) * 2 + e + g * 8));
                        f[v * 2 + e] = (_Float16)Wl[k];
                    }
                }
                wfrag[l][kf] = f;
            }
        }
    }
    // biases for owned column, all hidden layers
    float bh_reg[NLAYER];
    #pragma unroll
    for (int l = 0; l < NLAYER; ++l) bh_reg[l] = b_h[l * HIDDEN + col];

    // layer-1 weights for this thread's hidden unit (n = tid & 255)
    const int n1 = tid & 255;
    const int mh = tid >> 8;                 // 0/1 -> which 8 rows this thread fills
    float win_reg[5];
    #pragma unroll
    for (int i = 0; i < 5; ++i) win_reg[i] = W_in[n1 * 5 + i];
    const float bin_reg = b_in[n1];
    const float bo0 = b_out[0], bo1 = b_out[1];

    // output-layer weights: lane covers k = lane + 32*i, both output units
    float wo0[8], wo1[8];
    #pragma unroll
    for (int i = 0; i < 8; ++i) {
        wo0[i] = W_out[lane + i * 32];
        wo1[i] = W_out[HIDDEN + lane + i * 32];
    }

    // ---------------- per-row ODE state (driver lanes tid < 16) ----------------
    float zs0 = 0.f, zs1 = 0.f, tt = 0.f;
    if (tid < MT) {
        zs0 = z0[(row0 + tid) * 2 + 0];
        zs1 = z0[(row0 + tid) * 2 + 1];
        tt  = t0[row0 + tid];
    }

    for (int s = 0; s < steps; ++s) {
        float za0 = 0.f, za1 = 0.f;          // RK4 weighted k accumulator

        #pragma unroll 1
        for (int stage = 0; stage < 4; ++stage) {
            // ---- stage input (driver lanes) ----
            if (tid < MT) {
                float zi0 = zs0, zi1 = zs1, tin = tt;
                if (stage > 0) {
                    const float k0 = kbuf[tid][0], k1 = kbuf[tid][1];
                    const float w  = (stage == 1) ? 1.0f : 2.0f;   // k1:1, k2:2, k3:2
                    za0 += w * k0; za1 += w * k1;
                    const float c = (stage == 3) ? DT : 0.5f * DT;
                    zi0 = zs0 + c * k0;
                    zi1 = zs1 + c * k1;
                    tin = tt + c;
                }
                xbuf[tid][0] = tin;
                xbuf[tid][1] = zi0;
                xbuf[tid][2] = zi1;
                xbuf[tid][3] = __sinf(tin);      // FORCE_FREQ == 1
                xbuf[tid][4] = __cosf(tin);
            }
            __syncthreads();

            // ---- layer 1: 5 -> 256, scalar FMA, write hbuf[0] as f16 ----
            {
                #pragma unroll
                for (int mi = 0; mi < 8; ++mi) {
                    const int m = mh * 8 + mi;
                    float acc = bin_reg;
                    #pragma unroll
                    for (int i = 0; i < 5; ++i) acc += xbuf[m][i] * win_reg[i];
                    hbuf[0][m][n1] = (_Float16)fast_tanh(acc);
                }
            }
            __syncthreads();

            // ---- hidden layers: WMMA f16 -> f32 accum, tanh, ping-pong LDS ----
            int cur = 0;
            #pragma unroll
            for (int l = 0; l < NLAYER; ++l) {
                // Bias folded into C: D-layout gives one column per lane, so all
                // 8 acc elements share this lane's per-column bias.
                const float bias = bh_reg[l];
                v8f acc = {bias, bias, bias, bias, bias, bias, bias, bias};
                // Single accumulator: WMMA C-chaining has no documented hazard
                // (only D->A/B RAW needs NOPs), so back-to-back accumulation
                // pipelines and we save 8 VGPRs vs a split accumulator.
                #pragma unroll
                for (int kf = 0; kf < 8; ++kf) {
                    const h8 alo = *(const h8*)&hbuf[cur][nloc][kf * 32 + g * 8];
                    const h8 ahi = *(const h8*)&hbuf[cur][nloc][kf * 32 + 16 + g * 8];
                    const v16h a = __builtin_shufflevector(alo, ahi,
                        0,1,2,3,4,5,6,7,8,9,10,11,12,13,14,15);
                    acc = __builtin_amdgcn_wmma_f32_16x16x32_f16(
                        false, a, false, wfrag[l][kf], (short)0, acc, false, false);
                }
                // D layout: lanes 0-15 -> rows 0..7, lanes 16-31 -> rows 8..15
                #pragma unroll
                for (int v = 0; v < 8; ++v) {
                    const float hv = fast_tanh(acc[v]);
                    hbuf[cur ^ 1][g * 8 + v][col] = (_Float16)hv;
                }
                __syncthreads();
                cur ^= 1;
            }

            // ---- output layer: 256 -> 2, wave reduction per row ----
            {
                const int m = tid >> 5;      // 16 rows, one wave each
                float p0 = 0.f, p1 = 0.f;
                #pragma unroll
                for (int i = 0; i < 8; ++i) {
                    const float hk = (float)hbuf[cur][m][lane + i * 32];
                    p0 += hk * wo0[i];
                    p1 += hk * wo1[i];
                }
                #pragma unroll
                for (int off = 16; off > 0; off >>= 1) {
                    p0 += __shfl_xor(p0, off, 32);
                    p1 += __shfl_xor(p1, off, 32);
                }
                if (lane == 0) {
                    kbuf[m][0] = p0 + bo0;
                    kbuf[m][1] = p1 + bo1;
                }
            }
            __syncthreads();
        }

        // ---- RK4 update + store z_{s+1} ----
        if (tid < MT) {
            za0 += kbuf[tid][0];             // k4, weight 1
            za1 += kbuf[tid][1];
            zs0 += (DT / 6.0f) * za0;
            zs1 += (DT / 6.0f) * za1;
            tt  += DT;
            const size_t o = ((size_t)(row0 + tid) * (size_t)steps + (size_t)s) * 2;
            out[o + 0] = zs0;
            out[o + 1] = zs1;
        }
        __syncthreads();
    }
}

extern "C" void kernel_launch(void* const* d_in, const int* in_sizes, int n_in,
                              void* d_out, int out_size, void* d_ws, size_t ws_size,
                              hipStream_t stream) {
    const float* z0    = (const float*)d_in[0];
    const float* t0    = (const float*)d_in[1];
    const float* W_in  = (const float*)d_in[2];
    const float* b_in  = (const float*)d_in[3];
    const float* W_h   = (const float*)d_in[4];
    const float* b_h   = (const float*)d_in[5];
    const float* W_out = (const float*)d_in[6];
    const float* b_out = (const float*)d_in[7];
    const int*   steps = (const int*)d_in[8];

    const int B    = in_sizes[0] / 2;        // 4096
    const int nblk = B / MT;                 // 256 persistent workgroups

    hipLaunchKernelGGL(node_rollout_kernel, dim3(nblk), dim3(THREADS), 0, stream,
                       z0, t0, W_in, b_in, W_h, b_h, W_out, b_out, steps,
                       (float*)d_out);
}
